// CrossAttentionPairBias_60146722013687
// MI455X (gfx1250) — matrix operand order repaired and checked by reading, not compiled, not run on hardware
//
#include <hip/hip_runtime.h>
#include <stdint.h>

// ---- problem constants ----
#define B_   2
#define N_   384
#define H_   8
#define HM_  8
#define HS_  16
#define CS_  64
#define CZ_  128
#define Dh   160          // padded head dim (144 valid: 128 mv + 16 s)
#define DV_  144
#define NKP  416          // padded key count for transposed V
#define SCALE_ (1.0f/12.0f)   // 1/sqrt(16*HM + HS) = 1/sqrt(144)
#define INF_   100000.0f
#define EPS_   1e-5f
#define ROPEB_ 4096.0f

typedef _Float16 v16h __attribute__((ext_vector_type(16)));
typedef float    v8f  __attribute__((ext_vector_type(8)));

union ABreg { v16h v; uint32_t u[8]; };

// A-operand (16x32 f16) K base per (vgpr r, lane group g): ISA 7.12.2 table
__device__ __forceinline__ int kbA(int r, int g) {
    return (r < 4 ? 2 * r : 16 + 2 * (r - 4)) + 8 * g;
}

// ---- DPP16 cross-lane reductions over 16-lane halves (VALU, no DS) ----
// xor masks {1,2,7,15}: quad_perm(1,0,3,2)=0xB1, quad_perm(2,3,0,1)=0x4E,
// row_half_mirror=0x141 (j^7), row_mirror=0x140 (j^15)
template <int CTRL>
__device__ __forceinline__ float dpp_mov(float v) {
    return __builtin_bit_cast(
        float, __builtin_amdgcn_update_dpp(
                   0, __builtin_bit_cast(int, v), CTRL, 0xF, 0xF, true));
}
__device__ __forceinline__ float rowmax16(float v) {
    v = fmaxf(v, dpp_mov<0xB1>(v));
    v = fmaxf(v, dpp_mov<0x4E>(v));
    v = fmaxf(v, dpp_mov<0x141>(v));
    v = fmaxf(v, dpp_mov<0x140>(v));
    return v;
}
__device__ __forceinline__ float rowsum16(float v) {
    v += dpp_mov<0xB1>(v);
    v += dpp_mov<0x4E>(v);
    v += dpp_mov<0x141>(v);
    v += dpp_mov<0x140>(v);
    return v;
}

// block-wide sum over 256 slots (callers pad with zeros)
__device__ __forceinline__ float blk_reduce(float x, float* red, int t) {
    red[t] = x; __syncthreads();
    #pragma unroll
    for (int s = 128; s > 0; s >>= 1) {
        if (t < s) red[t] += red[t + s];
        __syncthreads();
    }
    float r = red[0]; __syncthreads();
    return r;
}

// ---------------------------------------------------------------------------
// Kernel Z: zero pad columns [384,416) of Vt for every (b,h,d) row
// ---------------------------------------------------------------------------
__global__ void zpad_kernel(_Float16* __restrict__ Vt) {
    int idx = blockIdx.x * 256 + threadIdx.x;     // B*H*Dh*32 = 81920
    if (idx >= B_ * H_ * Dh * 32) return;
    int col  = N_ + (idx & 31);
    int rest = idx >> 5;                          // bh*Dh + d
    Vt[(size_t)rest * NKP + col] = (_Float16)0.f;
}

// ---------------------------------------------------------------------------
// Kernel A: per-token pln + Q/KV projections + RoPE, pack per-head f16 rows
// Q,K: [bh][n][Dh] row-major.  Vt: [bh][Dh][NKP] (transposed for WMMA B).
// ---------------------------------------------------------------------------
__global__ void __launch_bounds__(256)
proj_kernel(const float* __restrict__ mv,  const float* __restrict__ sc,
            const float* __restrict__ Wq_mv, const float* __restrict__ Wq_s,
            const float* __restrict__ Wkv_mv, const float* __restrict__ Wkv_s,
            _Float16* __restrict__ Q, _Float16* __restrict__ K,
            _Float16* __restrict__ Vt) {
    __shared__ float smv[256], ssc[64], red[256];
    const int t  = threadIdx.x;
    const int tk = blockIdx.x;             // token: b*N + n
    const int b  = tk / N_, n = tk % N_;

    // pln over 256 multivector components
    float v = mv[(size_t)tk * 256 + t];
    float s1 = blk_reduce(v, red, t);
    float s2 = blk_reduce(v * v, red, t);
    float mean = s1 / 256.f;
    float var  = s2 / 256.f - mean * mean;
    smv[t] = (v - mean) * rsqrtf(var + EPS_);

    // pln over 64 scalars
    float sv = (t < 64) ? sc[(size_t)tk * 64 + t] : 0.f;
    float t1 = blk_reduce(sv, red, t);
    float t2 = blk_reduce(sv * sv, red, t);
    float ms = t1 / 64.f;
    float vs = t2 / 64.f - ms * ms;
    if (t < 64) ssc[t] = (sv - ms) * rsqrtf(vs + EPS_);
    __syncthreads();

    // ---- q_mv: o = hm*H + h ----
    for (int idx = t; idx < 64 * 16; idx += 256) {
        int o = idx >> 4, i = idx & 15;
        float acc = 0.f;
        #pragma unroll
        for (int c = 0; c < 16; ++c) acc += smv[c * 16 + i] * Wq_mv[o * 16 + c];
        int h = o & 7, hm = o >> 3;
        Q[(((size_t)b * H_ + h) * N_ + n) * Dh + hm * 16 + i] = (_Float16)acc;
    }
    // ---- kv_mv: o = kv*64 + hm*8 + h ----
    for (int idx = t; idx < 128 * 16; idx += 256) {
        int o = idx >> 4, i = idx & 15;
        float acc = 0.f;
        #pragma unroll
        for (int c = 0; c < 16; ++c) acc += smv[c * 16 + i] * Wkv_mv[o * 16 + c];
        int kv = o >> 6, oo = o & 63, hm = oo >> 3, h = oo & 7;
        if (kv == 0)
            K[(((size_t)b * H_ + h) * N_ + n) * Dh + hm * 16 + i] = (_Float16)acc;
        else
            Vt[(((size_t)b * H_ + h) * Dh + hm * 16 + i) * NKP + n] = (_Float16)acc;
    }
    // ---- q_s with RoPE: o = hs*8 + h, pairs hs=2j,2j+1 ----
    for (int idx = t; idx < 64; idx += 256) {
        int h = idx & 7, j = idx >> 3;
        float x1 = 0.f, x2 = 0.f;
        #pragma unroll
        for (int s = 0; s < CS_; ++s) {
            x1 += ssc[s] * Wq_s[((2 * j    ) * 8 + h) * CS_ + s];
            x2 += ssc[s] * Wq_s[((2 * j + 1) * 8 + h) * CS_ + s];
        }
        float inv = powf(ROPEB_, -(float)(2 * j) / 16.f);
        float ang = (float)n * inv, cs = cosf(ang), sn = sinf(ang);
        size_t base = (((size_t)b * H_ + h) * N_ + n) * Dh + 128;
        Q[base + 2 * j]     = (_Float16)(x1 * cs - x2 * sn);
        Q[base + 2 * j + 1] = (_Float16)(x1 * sn + x2 * cs);
    }
    // ---- kv_s: o = kv*128 + hs*8 + h; k gets RoPE, v is plain ----
    for (int idx = t; idx < 192; idx += 256) {
        if (idx < 64) {                    // k_s with RoPE
            int h = idx & 7, j = idx >> 3;
            float x1 = 0.f, x2 = 0.f;
            #pragma unroll
            for (int s = 0; s < CS_; ++s) {
                x1 += ssc[s] * Wkv_s[((2 * j    ) * 8 + h) * CS_ + s];
                x2 += ssc[s] * Wkv_s[((2 * j + 1) * 8 + h) * CS_ + s];
            }
            float inv = powf(ROPEB_, -(float)(2 * j) / 16.f);
            float ang = (float)n * inv, cs = cosf(ang), sn = sinf(ang);
            size_t base = (((size_t)b * H_ + h) * N_ + n) * Dh + 128;
            K[base + 2 * j]     = (_Float16)(x1 * cs - x2 * sn);
            K[base + 2 * j + 1] = (_Float16)(x1 * sn + x2 * cs);
        } else {                           // v_s
            int vidx = idx - 64;           // 0..127
            int hs = vidx >> 3, h = vidx & 7;
            int o = 128 + hs * 8 + h;
            float acc = 0.f;
            #pragma unroll
            for (int s = 0; s < CS_; ++s) acc += ssc[s] * Wkv_s[o * CS_ + s];
            Vt[(((size_t)b * H_ + h) * Dh + 128 + hs) * NKP + n] = (_Float16)acc;
        }
    }
    // ---- zero pad d = 144..159 ----
    for (int idx = t; idx < 128; idx += 256) {
        int h = idx >> 4, dd = idx & 15;
        size_t base = (((size_t)b * H_ + h) * N_ + n) * Dh + DV_ + dd;
        Q[base] = (_Float16)0.f;
        K[base] = (_Float16)0.f;
        Vt[(((size_t)b * H_ + h) * Dh + DV_ + dd) * NKP + n] = (_Float16)0.f;
    }
}

// ---------------------------------------------------------------------------
// Kernel B: bias[b,h,q,k] = <ln(pair[b,q,k,:]), Wpb[h,:]> + (1-mask)*INF
// One wave per 16-pair tile. LN stats: per-lane sums over 64 regs + one
// shfl_xor(16). Head projection: Z_norm(16x128) @ Wpb^T(128x16, heads in
// cols 0..7) as 4x wmma_f32_16x16x32_f16. This kernel streams the dominant
// 151 MB, so keep its non-VMEM work minimal.
// ---------------------------------------------------------------------------
__global__ void __launch_bounds__(256)
bias_kernel(const float* __restrict__ pair, const unsigned char* __restrict__ mask,
            const float* __restrict__ gamma, const float* __restrict__ beta,
            const float* __restrict__ Wpb, float* __restrict__ biasb) {
    const int lane = threadIdx.x & 31;
    const int w    = threadIdx.x >> 5;
    const int n16  = lane & 15, g = lane >> 4;
    long tile  = (long)blockIdx.x * 8 + w;        // B*N*N/16 = 18432 tiles
    if (tile >= (long)B_ * N_ * N_ / 16) return;
    long pbase = tile * 16;                       // 16 pairs: same (b,q), k0..k0+15

    // lane covers A-matrix row m = n16 (pair pbase+n16), K-half per group g:
    // chunk tt holds z = tt*32 + 8g + {0..7} and tt*32 + 16 + 8g + {0..7}
    const float* zp = pair + (pbase + n16) * CZ_;
    float raw[64];
    #pragma unroll
    for (int tt = 0; tt < 4; ++tt)
        #pragma unroll
        for (int hf = 0; hf < 2; ++hf) {
            const float* src = zp + tt * 32 + hf * 16 + 8 * g;
            #pragma unroll
            for (int e = 0; e < 8; ++e)
                raw[tt * 16 + hf * 8 + e] = src[e];
        }
    float s = 0.f, sq = 0.f;
    #pragma unroll
    for (int i = 0; i < 64; ++i) { s += raw[i]; sq += raw[i] * raw[i]; }
    s  += __shfl_xor(s, 16, 32);                  // combine the two z-halves
    sq += __shfl_xor(sq, 16, 32);
    float mean = s / 128.f;
    float rs   = rsqrtf(sq / 128.f - mean * mean + EPS_);

    // normalize + gamma/beta, pack A operands (element order == raw order)
    ABreg Az[4];
    #pragma unroll
    for (int tt = 0; tt < 4; ++tt)
        #pragma unroll
        for (int e = 0; e < 16; ++e) {
            int z = tt * 32 + 8 * g + (e < 8 ? e : e + 8);
            Az[tt].v[e] =
                (_Float16)((raw[tt * 16 + e] - mean) * rs * gamma[z] + beta[z]);
        }
    // B operand: Wpb^T (z x 16), cols 0..7 = heads, 8..15 zero
    ABreg Bw[4];
    #pragma unroll
    for (int tt = 0; tt < 4; ++tt)
        #pragma unroll
        for (int r = 0; r < 8; ++r) {
            int z0 = tt * 32 + kbA(r, g);
            float f0 = (n16 < 8) ? Wpb[n16 * CZ_ + z0]     : 0.f;
            float f1 = (n16 < 8) ? Wpb[n16 * CZ_ + z0 + 1] : 0.f;
            Bw[tt].v[2 * r]     = (_Float16)f0;
            Bw[tt].v[2 * r + 1] = (_Float16)f1;
        }
    v8f acc;
    #pragma unroll
    for (int e = 0; e < 8; ++e) acc[e] = 0.f;
    #pragma unroll
    for (int tt = 0; tt < 4; ++tt)
        acc = __builtin_amdgcn_wmma_f32_16x16x32_f16(
            false, Az[tt].v, false, Bw[tt].v, (short)0, acc, false, false);

    // D: lane col n16 = head, VGPR r -> pair m = r + 8g
    long b   = pbase / ((long)N_ * N_);
    long rem = pbase % ((long)N_ * N_);
    long q   = rem / N_, k0 = rem % N_;
    #pragma unroll
    for (int r = 0; r < 8; ++r) {
        int p = r + 8 * g;
        float mterm = (1.f - (float)mask[pbase + p]) * INF_;
        if (n16 < 8)
            biasb[((b * H_ + n16) * N_ + q) * N_ + k0 + p] = acc[r] + mterm;
    }
}

// ---------------------------------------------------------------------------
// Kernel C: flash attention, one wave per 16-row Q tile per (b,h).
// S = Q K^T via 5x wmma_f32_16x16x32_f16; online softmax in D-layout with
// DPP16 row reductions; O += P V via 10x WMMA (P transposed D->A through
// per-wave LDS).
// ---------------------------------------------------------------------------
__global__ void __launch_bounds__(128)
attn_kernel(const _Float16* __restrict__ Qg, const _Float16* __restrict__ Kg,
            const _Float16* __restrict__ Vt, const float* __restrict__ biasb,
            float* __restrict__ Hbuf) {
    __shared__ _Float16 Pls[4][16 * 16];     // per-wave P tile
    const int lane = threadIdx.x & 31;
    const int w    = threadIdx.x >> 5;
    const int tid  = blockIdx.x * 4 + w;     // 384 tiles total
    if (tid >= B_ * H_ * (N_ / 16)) return;
    const int bh = tid / (N_ / 16);
    const int q0 = (tid % (N_ / 16)) * 16;
    const int n16 = lane & 15, g = lane >> 4;

    // Q A-operands: row m = lane%16 (both halves), K interleave per kbA
    ABreg Aq[5];
    {
        const uint32_t* qrow =
            (const uint32_t*)(Qg + ((size_t)bh * N_ + q0 + n16) * Dh);
        #pragma unroll
        for (int tt = 0; tt < 5; ++tt)
            #pragma unroll
            for (int r = 0; r < 8; ++r)
                Aq[tt].u[r] = qrow[(tt * 32 + kbA(r, g)) >> 1];
    }

    v8f O[10];
    #pragma unroll
    for (int u = 0; u < 10; ++u)
        #pragma unroll
        for (int e = 0; e < 8; ++e) O[u][e] = 0.f;
    float Mrow[8], Lrow[8];
    #pragma unroll
    for (int r = 0; r < 8; ++r) { Mrow[r] = -1e30f; Lrow[r] = 0.f; }

    _Float16* Pw = Pls[w];

    for (int k0 = 0; k0 < N_; k0 += 16) {
        // ---- scores tile: S = Q K^T ----
        v8f Sacc;
        #pragma unroll
        for (int e = 0; e < 8; ++e) Sacc[e] = 0.f;
        const uint32_t* krow =
            (const uint32_t*)(Kg + ((size_t)bh * N_ + k0 + n16) * Dh);
        #pragma unroll
        for (int tt = 0; tt < 5; ++tt) {
            ABreg Bk;
            #pragma unroll
            for (int r = 0; r < 8; ++r)
                Bk.u[r] = krow[(tt * 32 + kbA(r, g)) >> 1];    // B layout: K=kbA(r,g)
            Sacc = __builtin_amdgcn_wmma_f32_16x16x32_f16(
                false, Aq[tt].v, false, Bk.v, (short)0, Sacc, false, false);
        }

        // ---- bias + online softmax (row m = r + 8g, col n16) ----
        const float* brow = biasb + ((size_t)bh * N_ + q0) * N_ + k0 + n16;
        float alpha[8];
        #pragma unroll
        for (int r = 0; r < 8; ++r) {
            int m = r + 8 * g;
            float sv = Sacc[r] * SCALE_ + brow[(size_t)m * N_];
            float mn = fmaxf(Mrow[r], rowmax16(sv));
            float a  = __expf(Mrow[r] - mn);
            float p  = __expf(sv - mn);
            Lrow[r] = Lrow[r] * a + rowsum16(p);
            Mrow[r] = mn;
            alpha[r] = a;
            Pw[m * 16 + n16] = (_Float16)p;    // D-layout scatter into LDS
        }
        // wave-private LDS turnaround: order the gather after the scatter
        asm volatile("s_wait_dscnt 0" ::: "memory");

        // P as A-operand (K = 0..15 real, 16..31 zero-padded)
        ABreg Ap;
        #pragma unroll
        for (int r = 0; r < 8; ++r) {
            int kb = kbA(r, g);
            Ap.u[r] = (kb < 16) ? *(const uint32_t*)(Pw + n16 * 16 + kb) : 0u;
        }

        // ---- O = O*alpha + P V ----
        #pragma unroll
        for (int u = 0; u < 10; ++u) {
            #pragma unroll
            for (int e = 0; e < 8; ++e) O[u][e] *= alpha[e];
            const uint32_t* vrow =
                (const uint32_t*)(Vt + ((size_t)bh * Dh + u * 16 + n16) * NKP);
            ABreg Bv;
            #pragma unroll
            for (int r = 0; r < 8; ++r)
                Bv.u[r] = vrow[(k0 + kbA(r, g)) >> 1];
            O[u] = __builtin_amdgcn_wmma_f32_16x16x32_f16(
                false, Ap.v, false, Bv.v, (short)0, O[u], false, false);
        }
    }

    // ---- normalize and write H ----
    float* hrow = Hbuf + ((size_t)bh * N_ + q0) * Dh;
    #pragma unroll
    for (int u = 0; u < 10; ++u)
        #pragma unroll
        for (int r = 0; r < 8; ++r) {
            int m = r + 8 * g;
            hrow[(size_t)m * Dh + u * 16 + n16] = O[u][r] / Lrow[r];
        }
}

// ---------------------------------------------------------------------------
// Kernel D: per-token output projections
// ---------------------------------------------------------------------------
__global__ void __launch_bounds__(256)
out_kernel(const float* __restrict__ Hbuf, const float* __restrict__ Wo_mv,
           const float* __restrict__ Wo_s, float* __restrict__ out) {
    __shared__ float sh[H_ * DV_];           // 1152 floats
    const int t  = threadIdx.x;
    const int tk = blockIdx.x;
    const int b  = tk / N_, n = tk % N_;
    for (int idx = t; idx < H_ * DV_; idx += 256) {
        int h = idx / DV_, d = idx % DV_;
        sh[idx] = Hbuf[(((size_t)b * H_ + h) * N_ + n) * Dh + d];
    }
    __syncthreads();
    {   // out_mv[o][i], c = h*HM + hm
        int o = t >> 4, i = t & 15;
        float acc = 0.f;
        #pragma unroll
        for (int c = 0; c < 64; ++c) {
            int h = c >> 3, hm = c & 7;
            acc += sh[h * DV_ + hm * 16 + i] * Wo_mv[o * 64 + c];
        }
        out[(((size_t)b * N_ + n) * 16 + o) * 16 + i] = acc;
    }
    if (t < 64) {   // out_s[o], c = h*HS + hs
        float acc = 0.f;
        #pragma unroll
        for (int c = 0; c < 128; ++c) {
            int h = c >> 4, hs = c & 15;
            acc += sh[h * DV_ + 128 + hs] * Wo_s[t * 128 + c];
        }
        out[(size_t)B_ * N_ * 256 + ((size_t)b * N_ + n) * 64 + t] = acc;
    }
}

// ---------------------------------------------------------------------------
extern "C" void kernel_launch(void* const* d_in, const int* in_sizes, int n_in,
                              void* d_out, int out_size, void* d_ws, size_t ws_size,
                              hipStream_t stream) {
    const float* mv     = (const float*)d_in[0];
    const float* sc     = (const float*)d_in[1];
    const float* pair   = (const float*)d_in[2];
    const unsigned char* mask = (const unsigned char*)d_in[3];
    const float* Wq_mv  = (const float*)d_in[4];
    const float* Wq_s   = (const float*)d_in[5];
    const float* Wkv_mv = (const float*)d_in[6];
    const float* Wkv_s  = (const float*)d_in[7];
    const float* Wo_mv  = (const float*)d_in[8];
    const float* Wo_s   = (const float*)d_in[9];
    const float* gamma  = (const float*)d_in[10];
    const float* beta   = (const float*)d_in[11];
    const float* Wpb    = (const float*)d_in[12];

    _Float16* Q  = (_Float16*)d_ws;
    _Float16* K  = Q  + (size_t)B_ * H_ * N_ * Dh;
    _Float16* Vt = K  + (size_t)B_ * H_ * N_ * Dh;
    float* biasb = (float*)(Vt + (size_t)B_ * H_ * Dh * NKP);
    float* Hbuf  = biasb + (size_t)B_ * H_ * N_ * N_;

    zpad_kernel<<<(B_ * H_ * Dh * 32 + 255) / 256, 256, 0, stream>>>(Vt);
    proj_kernel<<<B_ * N_, 256, 0, stream>>>(mv, sc, Wq_mv, Wq_s, Wkv_mv, Wkv_s,
                                             Q, K, Vt);
    bias_kernel<<<(B_ * N_ * N_ / 16) / 8, 256, 0, stream>>>(pair, mask, gamma,
                                                             beta, Wpb, biasb);
    attn_kernel<<<(B_ * H_ * (N_ / 16)) / 4, 128, 0, stream>>>(Q, K, Vt, biasb,
                                                               Hbuf);
    out_kernel<<<B_ * N_, 256, 0, stream>>>(Hbuf, Wo_mv, Wo_s, (float*)d_out);
}
